// RNNAttention_4080218931879
// MI455X (gfx1250) — matrix operand order repaired and checked by reading, not compiled, not run on hardware
//
#include <hip/hip_runtime.h>
#include <hip/hip_bf16.h>

// ---------------------------------------------------------------------------
// RNN-attention forecaster for MI455X (gfx1250, wave32).
//  * f16 storage + v_wmma_f32_16x16x32_f16 (f32 accum) for all GEMMs
//  * Dual independent WMMA accumulation chains per tile (even/odd K-steps)
//    so B-fragment loads can stay in flight behind the other chain's WMMA
//    instead of serializing at s_wait_loadcnt 0 before every WMMA.
//  * LSTM cell state kept in f32 VGPRs; gates round-trip LDS in f16
//  * Weights pre-packed into per-lane WMMA B-fragment order (L2-resident).
//  * 16 prediction windows are LSTM-independent -> fully parallel scan;
//    only the rank-1-reduced attention tail is sequential over 16 steps.
// ---------------------------------------------------------------------------

typedef __attribute__((ext_vector_type(16))) _Float16 v16h;
typedef __attribute__((ext_vector_type(8)))  _Float16 v8h;
typedef __attribute__((ext_vector_type(8)))  float    v8f;

constexpr int B_   = 512;
constexpr int LP_  = 16;
constexpr int LH_  = 64;
constexpr int WIN_ = 65;
constexpr int DIN_ = 128;
constexpr int H_   = 256;
constexpr int G_   = 1024;
constexpr int TSEQ = 80;   // LP + LH

__device__ __forceinline__ float sigf(float x) { return 1.0f / (1.0f + __expf(-x)); }

__device__ __forceinline__ v8f wmma_f16(v16h a, v16h b, v8f c) {
    return __builtin_amdgcn_wmma_f32_16x16x32_f16(false, a, false, b, (short)0, c,
                                                  false, false);
}

// Gather a 16x32 f16 A-fragment from row-major LDS rows (doc 7.12.2 layout):
// lanes 0-15: M=lane,   halves -> K = {0..7, 16..23} + kt*32
// lanes16-31: M=lane-16,halves -> K = {8..15,24..31} + kt*32
__device__ __forceinline__ v16h ldsA(const _Float16* base, int lda, int kt, int lane) {
    int m  = lane & 15;
    int kb = kt * 32 + ((lane >> 4) << 3);
    const _Float16* p = base + m * lda + kb;
    v8h lo = *(const v8h*)(p);
    v8h hi = *(const v8h*)(p + 16);
    v16h r;
#pragma unroll
    for (int j = 0; j < 8; ++j) { r[j] = lo[j]; r[8 + j] = hi[j]; }
    return r;
}

// ---------------------------------------------------------------------------
// Pack W (row-major [N][ldw] f32, using columns 0..K-1) into per-lane WMMA
// B-fragments: tile (nt,kt) -> 32 lanes x 16 halves contiguous.
// B[k][n] = W[n][k].  lanes 0-15: n=lane, K={0..7,16..23}; lanes 16-31: n=lane-16, K={8..15,24..31}.
// ---------------------------------------------------------------------------
__global__ void k_pack_b(const float* __restrict__ W, _Float16* __restrict__ dst,
                         int KT, int ldw) {
    int tile = blockIdx.x;            // nt*KT + kt
    int nt = tile / KT, kt = tile % KT;
    int t = threadIdx.x, lane = t & 31, h = t >> 5;   // 512 threads = 32x16
    int n = nt * 16 + (lane & 15);
    int k = kt * 32 + (h & 7) + ((lane >> 4) << 3) + ((h >> 3) << 4);
    dst[((size_t)tile * 32 + lane) * 16 + h] = (_Float16)W[(size_t)n * ldw + k];
}

// ---------------------------------------------------------------------------
// Small precompute: fused biases, rank-1 folding of the yt branch of FF1,
// log-target, and the first LH columns of the output.
// ---------------------------------------------------------------------------
__global__ void k_prep(const float* __restrict__ tgt,
                       const float* __restrict__ bih0, const float* __restrict__ bhh0,
                       const float* __restrict__ bih1, const float* __restrict__ bhh1,
                       const float* __restrict__ Wff1, const float* __restrict__ bff1,
                       const float* __restrict__ Wlin1, const float* __restrict__ blin1,
                       float* __restrict__ b0g, float* __restrict__ b1g,
                       float* __restrict__ bfc, float* __restrict__ vg,
                       float* __restrict__ lt, float* __restrict__ out) {
    int t = threadIdx.x;
    for (int g = t; g < G_; g += 256) {
        b0g[g] = bih0[g] + bhh0[g];
        b1g[g] = bih1[g] + bhh1[g];
        const float* wr = Wff1 + (size_t)g * G_ + 768;
        float sc = 0.f, sv = 0.f;
        for (int h = 0; h < H_; ++h) { sc += wr[h] * blin1[h]; sv += wr[h] * Wlin1[h]; }
        bfc[g] = bff1[g] + sc;
        vg[g]  = sv;
    }
    for (int e = t; e < B_ * TSEQ; e += 256) {
        float v = __logf(tgt[e] + 1e-5f);
        lt[e] = v;
        int b = e / TSEQ, tt = e - b * TSEQ;
        if (tt < LH_) out[b * TSEQ + tt] = v;
    }
}

// ---------------------------------------------------------------------------
// XP0[(t*B + b)][g] = X[b,t,:] @ W_ih0.T + (b_ih0+b_hh0), stored f16.
// [t][b][g] layout so the LSTM's per-timestep reads stride by 2KB.
// ---------------------------------------------------------------------------
__global__ void __launch_bounds__(256) k_xp0(const float* __restrict__ X,
                                             const _Float16* __restrict__ Wb,
                                             const float* __restrict__ b0g,
                                             _Float16* __restrict__ XP) {
    __shared__ __align__(16) _Float16 sA[16 * 136];
    int R = blockIdx.x * 16;             // row = t*B + b
    int tt = R >> 9;                     // 512 rows per t; tiles never cross t
    int b0 = R & 511;
    int tid = threadIdx.x, lane = tid & 31, wv = tid >> 5;
    for (int e = tid; e < 16 * DIN_; e += 256) {
        int m = e >> 7, k = e & 127;
        sA[m * 136 + k] = (_Float16)X[((size_t)(b0 + m) * TSEQ + tt) * DIN_ + k];
    }
    __syncthreads();
    v16h a[4];
#pragma unroll
    for (int kt = 0; kt < 4; ++kt) a[kt] = ldsA(sA, 136, kt, lane);
    for (int j = 0; j < 8; ++j) {
        int nt = wv * 8 + j;
        int n  = nt * 16 + (lane & 15);
        float bv = b0g[n];
        v8f accA, accB;
#pragma unroll
        for (int r = 0; r < 8; ++r) { accA[r] = bv; accB[r] = 0.f; }
        v16h bm[4];
#pragma unroll
        for (int kt = 0; kt < 4; ++kt)
            bm[kt] = *(const v16h*)(Wb + ((size_t)(nt * 4 + kt) * 32 + lane) * 16);
        accA = wmma_f16(a[0], bm[0], accA);
        accB = wmma_f16(a[1], bm[1], accB);
        accA = wmma_f16(a[2], bm[2], accA);
        accB = wmma_f16(a[3], bm[3], accB);
#pragma unroll
        for (int r = 0; r < 8; ++r) {
            int m = r + ((lane >> 4) << 3);
            XP[(size_t)(R + m) * G_ + n] = (_Float16)(accA[r] + accB[r]);
        }
    }
}

// ---------------------------------------------------------------------------
// Fused 2-layer LSTM scan over 65 timesteps.
// Grid: (32 batch-tiles of 16) x (16 windows).  256 threads = 8 wave32.
// A-fragments hoisted per gemm phase; dual WMMA chains per tile.
// ---------------------------------------------------------------------------
__global__ void __launch_bounds__(256) k_lstm(const _Float16* __restrict__ XP,
                                              const _Float16* __restrict__ WbHH0,
                                              const _Float16* __restrict__ WbIH1,
                                              const _Float16* __restrict__ WbHH1,
                                              const float* __restrict__ b1g,
                                              _Float16* __restrict__ OUT,
                                              _Float16* __restrict__ HFIN) {
    __shared__ __align__(16) _Float16 hA0[16 * 272];
    __shared__ __align__(16) _Float16 hA1[16 * 272];
    __shared__ __align__(16) _Float16 gsh[16 * 1040];
    const int i   = blockIdx.y;
    const int bB  = blockIdx.x * 16;
    const int tid = threadIdx.x, lane = tid & 31, wv = tid >> 5;
    const int erow = tid >> 4;            // batch row handled in elementwise phase
    const int ehb  = (tid & 15) * 16;     // 16 hidden units per thread

    for (int e = tid; e < 16 * 272; e += 256) { hA0[e] = (_Float16)0.f; hA1[e] = (_Float16)0.f; }
    float c0[16], c1[16];
#pragma unroll
    for (int j = 0; j < 16; ++j) { c0[j] = 0.f; c1[j] = 0.f; }
    __syncthreads();

    for (int t = 0; t < WIN_; ++t) {
        // ---- layer-0 gates: g = XP[t] + h0 @ Whh0^T
        {
            v16h a0[8];
#pragma unroll
            for (int kt = 0; kt < 8; ++kt) a0[kt] = ldsA(hA0, 272, kt, lane);
            for (int j = 0; j < 8; ++j) {
                int nt = wv * 8 + j;
                int n  = nt * 16 + (lane & 15);
                v8f accA, accB;
#pragma unroll
                for (int r = 0; r < 8; ++r) {
                    int m = r + ((lane >> 4) << 3);
                    accA[r] = (float)XP[((size_t)(i + t) * B_ + (bB + m)) * G_ + n];
                    accB[r] = 0.f;
                }
                v16h bm[8];
#pragma unroll
                for (int kt = 0; kt < 8; ++kt)
                    bm[kt] = *(const v16h*)(WbHH0 + ((size_t)(nt * 8 + kt) * 32 + lane) * 16);
#pragma unroll
                for (int kt = 0; kt < 4; ++kt) {
                    accA = wmma_f16(a0[2 * kt],     bm[2 * kt],     accA);
                    accB = wmma_f16(a0[2 * kt + 1], bm[2 * kt + 1], accB);
                }
#pragma unroll
                for (int r = 0; r < 8; ++r) {
                    int m = r + ((lane >> 4) << 3);
                    gsh[m * 1040 + n] = (_Float16)(accA[r] + accB[r]);
                }
            }
        }
        __syncthreads();
        // ---- layer-0 cell update
        {
            int b = bB + erow;
#pragma unroll
            for (int j = 0; j < 16; ++j) {
                int h = ehb + j;
                float gi = (float)gsh[erow * 1040 + h];
                float gf = (float)gsh[erow * 1040 + h + 256];
                float gg = (float)gsh[erow * 1040 + h + 512];
                float go = (float)gsh[erow * 1040 + h + 768];
                float c  = sigf(gf) * c0[j] + sigf(gi) * tanhf(gg);
                c0[j] = c;
                float hv = sigf(go) * tanhf(c);
                hA0[erow * 272 + h] = (_Float16)hv;
                if (t == WIN_ - 1) HFIN[((size_t)i * B_ + b) * 512 + h] = (_Float16)hv;
            }
        }
        __syncthreads();
        // ---- layer-1 gates: g = h0(t) @ Wih1^T + h1(t-1) @ Whh1^T + b1
        {
            v16h a0[8], a1[8];
#pragma unroll
            for (int kt = 0; kt < 8; ++kt) a0[kt] = ldsA(hA0, 272, kt, lane);
#pragma unroll
            for (int kt = 0; kt < 8; ++kt) a1[kt] = ldsA(hA1, 272, kt, lane);
            for (int j = 0; j < 8; ++j) {
                int nt = wv * 8 + j;
                int n  = nt * 16 + (lane & 15);
                float bv = b1g[n];
                v8f accA, accB;
#pragma unroll
                for (int r = 0; r < 8; ++r) { accA[r] = bv; accB[r] = 0.f; }
                {
                    v16h bm[8];
#pragma unroll
                    for (int kt = 0; kt < 8; ++kt)
                        bm[kt] = *(const v16h*)(WbIH1 + ((size_t)(nt * 8 + kt) * 32 + lane) * 16);
#pragma unroll
                    for (int kt = 0; kt < 4; ++kt) {
                        accA = wmma_f16(a0[2 * kt],     bm[2 * kt],     accA);
                        accB = wmma_f16(a0[2 * kt + 1], bm[2 * kt + 1], accB);
                    }
                }
                {
                    v16h bm[8];
#pragma unroll
                    for (int kt = 0; kt < 8; ++kt)
                        bm[kt] = *(const v16h*)(WbHH1 + ((size_t)(nt * 8 + kt) * 32 + lane) * 16);
#pragma unroll
                    for (int kt = 0; kt < 4; ++kt) {
                        accA = wmma_f16(a1[2 * kt],     bm[2 * kt],     accA);
                        accB = wmma_f16(a1[2 * kt + 1], bm[2 * kt + 1], accB);
                    }
                }
#pragma unroll
                for (int r = 0; r < 8; ++r) {
                    int m = r + ((lane >> 4) << 3);
                    gsh[m * 1040 + n] = (_Float16)(accA[r] + accB[r]);
                }
            }
        }
        __syncthreads();
        // ---- layer-1 cell update
        {
            int b = bB + erow;
#pragma unroll
            for (int j = 0; j < 16; ++j) {
                int h = ehb + j;
                float gi = (float)gsh[erow * 1040 + h];
                float gf = (float)gsh[erow * 1040 + h + 256];
                float gg = (float)gsh[erow * 1040 + h + 512];
                float go = (float)gsh[erow * 1040 + h + 768];
                float c  = sigf(gf) * c1[j] + sigf(gi) * tanhf(gg);
                c1[j] = c;
                float hv = sigf(go) * tanhf(c);
                hA1[erow * 272 + h] = (_Float16)hv;
                if (t < LH_) OUT[(((size_t)i * B_ + b) * LH_ + t) * H_ + h] = (_Float16)hv;
                else         HFIN[((size_t)i * B_ + b) * 512 + 256 + h] = (_Float16)hv;
            }
        }
        __syncthreads();
    }
}

// ---------------------------------------------------------------------------
// PH[b][g] = ht_i[b,:] @ W_ff1[:,256:768].T  (K=512, packed FF1 buffer at
// K-tile offset 8).  A-fragments fully hoisted; dual WMMA chains.
// ---------------------------------------------------------------------------
__global__ void __launch_bounds__(256) k_preht(const _Float16* __restrict__ HFIN,
                                               const _Float16* __restrict__ WbFF1,
                                               float* __restrict__ PH, int i) {
    __shared__ __align__(16) _Float16 sA[16 * 520];
    int R = blockIdx.x * 16;
    int tid = threadIdx.x, lane = tid & 31, wv = tid >> 5;
    for (int e = tid; e < 16 * 512; e += 256) {
        int m = e >> 9, k = e & 511;
        sA[m * 520 + k] = HFIN[((size_t)i * B_ + (R + m)) * 512 + k];
    }
    __syncthreads();
    v16h a[16];
#pragma unroll
    for (int kt = 0; kt < 16; ++kt) a[kt] = ldsA(sA, 520, kt, lane);
    for (int j = 0; j < 8; ++j) {
        int nt = wv * 8 + j;
        int n  = nt * 16 + (lane & 15);
        v8f accA, accB;
#pragma unroll
        for (int r = 0; r < 8; ++r) { accA[r] = 0.f; accB[r] = 0.f; }
#pragma unroll
        for (int grp = 0; grp < 2; ++grp) {
            v16h bm[8];
#pragma unroll
            for (int kk = 0; kk < 8; ++kk)
                bm[kk] = *(const v16h*)(WbFF1 +
                         ((size_t)(nt * 24 + 8 + grp * 8 + kk) * 32 + lane) * 16);
#pragma unroll
            for (int kk = 0; kk < 4; ++kk) {
                accA = wmma_f16(a[grp * 8 + 2 * kk],     bm[2 * kk],     accA);
                accB = wmma_f16(a[grp * 8 + 2 * kk + 1], bm[2 * kk + 1], accB);
            }
        }
#pragma unroll
        for (int r = 0; r < 8; ++r) {
            int m = r + ((lane >> 4) << 3);
            PH[(size_t)(R + m) * G_ + n] = accA[r] + accB[r];
        }
    }
}

// ---------------------------------------------------------------------------
// Attention logits for step i:  rows = (b,l), 32768 of them.
//   acc = [out|ht] @ W_ff1[:, :768].T + bfc + PH[b] + yi(b,l)*v   (rank-1 yt)
//   logit[b,l] = sum_g tanh(acc)*W_lin2[g] + b_lin2
// K-group-major loop with 8 live accumulators (8 independent WMMA chains).
// ---------------------------------------------------------------------------
__global__ void __launch_bounds__(256) k_logits(const _Float16* __restrict__ OUT,
                                                const _Float16* __restrict__ HFIN,
                                                const _Float16* __restrict__ WbFF1,
                                                const float* __restrict__ PH,
                                                const float* __restrict__ bfc,
                                                const float* __restrict__ vg,
                                                const float* __restrict__ Wl2,
                                                const float* __restrict__ bl2,
                                                const float* __restrict__ lt,
                                                const float* __restrict__ preds,
                                                float* __restrict__ logits, int i) {
    __shared__ __align__(16) _Float16 sA[16 * 776];
    __shared__ float red[16][8];
    int R  = blockIdx.x * 16;        // row = b*64 + l
    int b  = R >> 6;
    int l0 = R & 63;
    int tid = threadIdx.x, lane = tid & 31, wv = tid >> 5;

    for (int e = tid; e < 16 * 768; e += 256) {
        int m = e / 768, k = e - m * 768;
        _Float16 v;
        if (k < 256) v = OUT[(((size_t)i * B_ + b) * LH_ + (l0 + m)) * H_ + k];
        else         v = HFIN[((size_t)i * B_ + b) * 512 + (k - 256)];
        sA[m * 776 + k] = v;
    }
    __syncthreads();

    float yv[8];
#pragma unroll
    for (int r = 0; r < 8; ++r) {
        int l = l0 + r + ((lane >> 4) << 3);
        yv[r] = (l < LH_ - i) ? lt[b * TSEQ + i + l] : preds[b * LP_ + (l - (LH_ - i))];
    }

    v8f acc[8];
#pragma unroll
    for (int j = 0; j < 8; ++j) {
        int nt = wv * 8 + j;
        int n  = nt * 16 + (lane & 15);
        float base = bfc[n] + PH[(size_t)b * G_ + n];
        float vn   = vg[n];
#pragma unroll
        for (int r = 0; r < 8; ++r) acc[j][r] = base + yv[r] * vn;
    }
    for (int grp = 0; grp < 3; ++grp) {
        v16h a[8];
#pragma unroll
        for (int kk = 0; kk < 8; ++kk) a[kk] = ldsA(sA, 776, grp * 8 + kk, lane);
        for (int j = 0; j < 8; ++j) {
            int nt = wv * 8 + j;
            v16h bm[8];
#pragma unroll
            for (int kk = 0; kk < 8; ++kk)
                bm[kk] = *(const v16h*)(WbFF1 +
                         ((size_t)(nt * 24 + grp * 8 + kk) * 32 + lane) * 16);
#pragma unroll
            for (int kk = 0; kk < 8; ++kk) acc[j] = wmma_f16(a[kk], bm[kk], acc[j]);
        }
    }
    float tsum[8];
#pragma unroll
    for (int r = 0; r < 8; ++r) tsum[r] = 0.f;
#pragma unroll
    for (int j = 0; j < 8; ++j) {
        int n  = (wv * 8 + j) * 16 + (lane & 15);
        float w2 = Wl2[n];
#pragma unroll
        for (int r = 0; r < 8; ++r) tsum[r] += tanhf(acc[j][r]) * w2;
    }
#pragma unroll
    for (int r = 0; r < 8; ++r) {
#pragma unroll
        for (int off = 8; off >= 1; off >>= 1)
            tsum[r] += __shfl_xor(tsum[r], off, 32);   // within 16-lane half
    }
    if ((lane & 15) == 0) {
#pragma unroll
        for (int r = 0; r < 8; ++r) {
            int m = r + ((lane >> 4) << 3);
            red[m][wv] = tsum[r];
        }
    }
    __syncthreads();
    if (tid < 16) {
        float s = bl2[0];
#pragma unroll
        for (int w = 0; w < 8; ++w) s += red[tid][w];
        logits[R + tid] = s;
    }
}

// ---------------------------------------------------------------------------
// Softmax over l, pooled state, FF2 -> pred[b,i] and output column 64+i.
// ---------------------------------------------------------------------------
__global__ void __launch_bounds__(256) k_pool(const float* __restrict__ logits,
                                              const _Float16* __restrict__ OUT,
                                              const float* __restrict__ Wa,
                                              const float* __restrict__ ba,
                                              const float* __restrict__ Wb,
                                              const float* __restrict__ bb,
                                              float* __restrict__ preds,
                                              float* __restrict__ out, int i) {
    __shared__ float ev[64];
    __shared__ float st[256];
    __shared__ float red[256];
    int b = blockIdx.x, tid = threadIdx.x;
    if (tid < LH_) ev[tid] = logits[b * LH_ + tid];
    __syncthreads();
    if (tid == 0) {
        float mx = ev[0];
        for (int l = 1; l < LH_; ++l) mx = fmaxf(mx, ev[l]);
        float s = 0.f;
        for (int l = 0; l < LH_; ++l) { float x = __expf(ev[l] - mx); ev[l] = x; s += x; }
        float inv = 1.f / s;
        for (int l = 0; l < LH_; ++l) ev[l] *= inv;
    }
    __syncthreads();
    {
        int h = tid;
        float s = 0.f;
        for (int l = 0; l < LH_; ++l)
            s += ev[l] * (float)OUT[(((size_t)i * B_ + b) * LH_ + l) * H_ + h];
        st[h] = s;
    }
    __syncthreads();
    {
        int jj = tid;
        float a = ba[jj];
        const float* wr = Wa + (size_t)jj * H_;
        for (int h = 0; h < H_; ++h) a += wr[h] * st[h];
        red[jj] = tanhf(a) * Wb[jj];
    }
    __syncthreads();
    for (int off = 128; off >= 1; off >>= 1) {
        if (tid < off) red[tid] += red[tid + off];
        __syncthreads();
    }
    if (tid == 0) {
        float p = red[0] + bb[0];
        preds[b * LP_ + i] = p;
        out[b * TSEQ + LH_ + i] = p;
    }
}

// ---------------------------------------------------------------------------
extern "C" void kernel_launch(void* const* d_in, const int* in_sizes, int n_in,
                              void* d_out, int out_size, void* d_ws, size_t ws_size,
                              hipStream_t stream) {
    (void)in_sizes; (void)n_in; (void)out_size; (void)ws_size;
    const float* X     = (const float*)d_in[0];
    const float* tgt   = (const float*)d_in[1];
    const float* Wih0  = (const float*)d_in[2];
    const float* Whh0  = (const float*)d_in[3];
    const float* bih0  = (const float*)d_in[4];
    const float* bhh0  = (const float*)d_in[5];
    const float* Wih1  = (const float*)d_in[6];
    const float* Whh1  = (const float*)d_in[7];
    const float* bih1  = (const float*)d_in[8];
    const float* bhh1  = (const float*)d_in[9];
    const float* Wlin1 = (const float*)d_in[10];
    const float* blin1 = (const float*)d_in[11];
    const float* Wff1  = (const float*)d_in[12];
    const float* bff1  = (const float*)d_in[13];
    const float* Wlin2 = (const float*)d_in[14];
    const float* blin2 = (const float*)d_in[15];
    const float* Wff2a = (const float*)d_in[16];
    const float* bff2a = (const float*)d_in[17];
    const float* Wff2b = (const float*)d_in[18];
    const float* bff2b = (const float*)d_in[19];
    float* out = (float*)d_out;

    char*  ws  = (char*)d_ws;
    size_t off = 0;
    auto alloc = [&](size_t bytes) -> char* {
        char* p = ws + off;
        off = (off + bytes + 255) & ~(size_t)255;
        return p;
    };
    _Float16* WbIH0 = (_Float16*)alloc((size_t)64 * 4  * 512 * 2);
    _Float16* WbHH0 = (_Float16*)alloc((size_t)64 * 8  * 512 * 2);
    _Float16* WbIH1 = (_Float16*)alloc((size_t)64 * 8  * 512 * 2);
    _Float16* WbHH1 = (_Float16*)alloc((size_t)64 * 8  * 512 * 2);
    _Float16* WbFF1 = (_Float16*)alloc((size_t)64 * 24 * 512 * 2);
    _Float16* XP    = (_Float16*)alloc((size_t)B_ * TSEQ * G_ * 2);
    _Float16* OUTb  = (_Float16*)alloc((size_t)LP_ * B_ * LH_ * H_ * 2);
    _Float16* HFIN  = (_Float16*)alloc((size_t)LP_ * B_ * 512 * 2);
    float* b0g   = (float*)alloc(G_ * 4);
    float* b1g   = (float*)alloc(G_ * 4);
    float* bfc   = (float*)alloc(G_ * 4);
    float* vg    = (float*)alloc(G_ * 4);
    float* lt    = (float*)alloc((size_t)B_ * TSEQ * 4);
    float* preds = (float*)alloc((size_t)B_ * LP_ * 4);
    float* lgts  = (float*)alloc((size_t)B_ * LH_ * 4);
    float* PH    = (float*)alloc((size_t)B_ * G_ * 4);

    k_prep<<<1, 256, 0, stream>>>(tgt, bih0, bhh0, bih1, bhh1, Wff1, bff1,
                                  Wlin1, blin1, b0g, b1g, bfc, vg, lt, out);
    k_pack_b<<<64 * 4,  512, 0, stream>>>(Wih0, WbIH0, 4, 128);
    k_pack_b<<<64 * 8,  512, 0, stream>>>(Whh0, WbHH0, 8, 256);
    k_pack_b<<<64 * 8,  512, 0, stream>>>(Wih1, WbIH1, 8, 256);
    k_pack_b<<<64 * 8,  512, 0, stream>>>(Whh1, WbHH1, 8, 256);
    k_pack_b<<<64 * 24, 512, 0, stream>>>(Wff1, WbFF1, 24, 1024);

    k_xp0<<<(B_ * TSEQ) / 16, 256, 0, stream>>>(X, WbIH0, b0g, XP);
    k_lstm<<<dim3(B_ / 16, LP_), 256, 0, stream>>>(XP, WbHH0, WbIH1, WbHH1,
                                                   b1g, OUTb, HFIN);
    for (int i = 0; i < LP_; ++i) {
        k_preht<<<B_ / 16, 256, 0, stream>>>(HFIN, WbFF1, PH, i);
        k_logits<<<(B_ * LH_) / 16, 256, 0, stream>>>(OUTb, HFIN, WbFF1, PH, bfc, vg,
                                                      Wlin2, blin2, lt, preds, lgts, i);
        k_pool<<<B_, 256, 0, stream>>>(lgts, OUTb, Wff2a, bff2a, Wff2b, bff2b,
                                       preds, out, i);
    }
}
